// FlashOccDepthViewTransformer_16097537425506
// MI455X (gfx1250) — matrix operand order repaired and checked by reading, not compiled, not run on hardware
//
#include <hip/hip_runtime.h>
#include <hip/hip_bf16.h>
#include <math.h>

typedef __attribute__((ext_vector_type(16))) _Float16 v16h;
typedef __attribute__((ext_vector_type(8)))  _Float16 v8h;
typedef __attribute__((ext_vector_type(8)))  float    v8f;

#define N_IMG  12
#define C_IN   256
#define HW     2816          // 32*88
#define P_TOT  33792         // N_IMG*HW
#define D_BINS 88
#define CC     128
#define O_OUT  216           // D_BINS + CC
#define O_PAD  224           // 14 tiles of 16
#define BEV_SZ (N_IMG * C_IN * HW)      // 8,650,752
#define LOG_SZ (N_IMG * D_BINS * HW)    // 2,973,696
#define BN_EPS 1e-5f

// ---------------------------------------------------------------------------
// Prep: f16-convert + zero-pad dnet_w -> wf16[224][256]; bias -> biasp[224];
//       proj_w -> pw16[256][128]; fold BN into scale/shift.
// ---------------------------------------------------------------------------
__global__ void k_prep(const float* __restrict__ w,  const float* __restrict__ b,
                       const float* __restrict__ pw, const float* __restrict__ g,
                       const float* __restrict__ bt, const float* __restrict__ mn,
                       const float* __restrict__ vr,
                       _Float16* __restrict__ wf16, float* __restrict__ biasp,
                       _Float16* __restrict__ pw16, float* __restrict__ bnsc,
                       float* __restrict__ bnsh)
{
    const int id   = blockIdx.x * 256 + threadIdx.x;
    const int step = gridDim.x * 256;
    for (int i = id; i < O_PAD * C_IN; i += step) {
        const int o = i >> 8, c = i & 255;
        wf16[i] = (o < O_OUT) ? (_Float16)w[o * C_IN + c] : (_Float16)0.0f;
    }
    for (int i = id; i < C_IN * CC; i += step) pw16[i] = (_Float16)pw[i];
    for (int i = id; i < O_PAD; i += step) biasp[i] = (i < O_OUT) ? b[i] : 0.0f;
    for (int i = id; i < C_IN; i += step) {
        const float sc = g[i] * rsqrtf(vr[i] + BN_EPS);
        bnsc[i] = sc;
        bnsh[i] = bt[i] - mn[i] * sc;
    }
}

// ---------------------------------------------------------------------------
// K1: dc[p][o] = sum_c x[p,c]*W[o,c] + b[o]   (f16 WMMA, f32 accum)
// One wave: one 16-position M-tile x ALL 14 O-tiles (x read exactly once).
// launch_bounds(128,1): no VGPR cap -> all accumulators stay in registers.
// B-fragments double-buffered (distance 1) to hide load latency behind WMMA.
// ---------------------------------------------------------------------------
__global__ void __launch_bounds__(128, 1)
k_gemm1(const float* __restrict__ x, const _Float16* __restrict__ wf16,
        const float* __restrict__ biasp,
        float* __restrict__ lg, _Float16* __restrict__ tf16)
{
    const int wid  = threadIdx.x >> 5;
    const int lane = threadIdx.x & 31;
    const int half = lane >> 4;
    const int l16  = lane & 15;

    const int mtile = blockIdx.x * 4 + wid;     // 0..2111
    const int pbase = mtile * 16;
    const int m     = pbase + l16;              // A row for this lane
    const int nimg  = m / HW;
    const int hw    = m % HW;

    // A layout: a[i] = A[m][kb + 16*half + i] ; half*16 folded in
    const float* xp = x + nimg * C_IN * HW + hw + (half << 4) * HW;
    // B layout: b[i] = B[kb + 16*half + i][o],  o = t*16 + l16
    const _Float16* wfp = wf16 + l16 * C_IN + (half << 4);

    v8f acc[14];
#pragma unroll
    for (int t = 0; t < 14; ++t) {
        const float bv = biasp[t * 16 + l16];
#pragma unroll
        for (int j = 0; j < 8; ++j) acc[t][j] = bv;
    }

#pragma unroll 1
    for (int kb = 0; kb < C_IN; kb += 32) {
        v16h a;
#pragma unroll
        for (int i = 0; i < 16; ++i)
            a[i] = (_Float16)xp[i * HW];  // imm offsets
        xp += 32 * HW;

        v16h bf = *(const v16h*)(wfp + kb);                         // tile 0
#pragma unroll
        for (int t = 0; t < 14; ++t) {
            v16h bfn = bf;
            if (t < 13)                                             // prefetch t+1
                bfn = *(const v16h*)(wfp + (t + 1) * 16 * C_IN + kb);
            acc[t] = __builtin_amdgcn_wmma_f32_16x16x32_f16(
                false, a, false, bf, (short)0, acc[t], false, false);
            bf = bfn;
        }
    }

    // D layout: VGPR j -> row m = pbase + j + 8*half, col o = l16
#pragma unroll
    for (int t = 0; t < 14; ++t) {
        const int o = t * 16 + l16;
#pragma unroll
        for (int j = 0; j < 8; ++j) {
            const int mm = pbase + j + (half << 3);
            const float v = acc[t][j];
            if (o < D_BINS)      lg[mm * D_BINS + o] = v;
            else if (o < O_OUT)  tf16[mm * CC + (o - D_BINS)] = (_Float16)v;
        }
    }
}

// ---------------------------------------------------------------------------
// K2: softmax over lg[p][0..87]; write logits + prob outputs; s[p].
// One wave per position, 8 positions per 256-thread block.
// ---------------------------------------------------------------------------
__device__ __forceinline__ float depth_weight(int d)
{
    return (1.0f + 44.0f * (float)d / 87.0f) * (1.0f / 23.0f);  // linspace/mean(=23)
}

__global__ void k_depth(const float* __restrict__ lg, float* __restrict__ out,
                        float* __restrict__ s)
{
    const int wid  = threadIdx.x >> 5;
    const int lane = threadIdx.x & 31;
    const int p    = blockIdx.x * 8 + wid;
    const int nimg = p / HW;
    const int hw   = p % HW;

    const float* row = lg + p * D_BINS;
    const float x0 = row[lane];
    const float x1 = row[lane + 32];
    const float x2 = (lane < 24) ? row[lane + 64] : -1e30f;

    float mx = fmaxf(fmaxf(x0, x1), x2);
#pragma unroll
    for (int off = 16; off >= 1; off >>= 1)
        mx = fmaxf(mx, __shfl_xor(mx, off, 32));

    const float e0 = expf(x0 - mx);
    const float e1 = expf(x1 - mx);
    const float e2 = (lane < 24) ? expf(x2 - mx) : 0.0f;

    float sum = e0 + e1 + e2;
    float sw  = e0 * depth_weight(lane) + e1 * depth_weight(lane + 32)
              + ((lane < 24) ? e2 * depth_weight(lane + 64) : 0.0f);
#pragma unroll
    for (int off = 16; off >= 1; off >>= 1) {
        sum += __shfl_xor(sum, off, 32);
        sw  += __shfl_xor(sw, off, 32);
    }
    const float inv = 1.0f / sum;

    float* lo = out + BEV_SZ          + (nimg * D_BINS) * HW + hw;
    float* pr = out + BEV_SZ + LOG_SZ + (nimg * D_BINS) * HW + hw;
    lo[lane * HW]        = x0;
    lo[(lane + 32) * HW] = x1;
    pr[lane * HW]        = e0 * inv;
    pr[(lane + 32) * HW] = e1 * inv;
    if (lane < 24) {
        lo[(lane + 64) * HW] = x2;
        pr[(lane + 64) * HW] = e2 * inv;
    }
    if (lane == 0) s[p] = sw * inv;
}

// ---------------------------------------------------------------------------
// K3: tf[c][p] = sum_k proj_w[c,k]*tf16[p][k]; BN + exact GELU + *s[p] -> bev.
// M = channel c (rows), N = position p (lanes) -> coalesced bev stores.
// One wave: one 16-position tile x ALL 16 C-tiles (tf16 read exactly once).
// A-fragments double-buffered (distance 1).
// ---------------------------------------------------------------------------
__global__ void __launch_bounds__(128, 1)
k_gemm2(const _Float16* __restrict__ tf16, const _Float16* __restrict__ pw16,
        const float* __restrict__ bnsc, const float* __restrict__ bnsh,
        const float* __restrict__ s,    float* __restrict__ out)
{
    const int wid  = threadIdx.x >> 5;
    const int lane = threadIdx.x & 31;
    const int half = lane >> 4;
    const int l16  = lane & 15;

    const int ptile = blockIdx.x * 4 + wid;     // 0..2111
    const int pbase = ptile * 16;
    const int p     = pbase + l16;              // B column (position)

    const _Float16* tfp = tf16 + p * CC + (half << 4);          // B-frag base
    const _Float16* pwp = pw16 + l16 * CC + (half << 4);        // A-frag base

    v8f acc[16];
#pragma unroll
    for (int t = 0; t < 16; ++t)
#pragma unroll
        for (int j = 0; j < 8; ++j) acc[t][j] = 0.0f;

#pragma unroll 1
    for (int kb = 0; kb < CC; kb += 32) {
        const v16h bf = *(const v16h*)(tfp + kb);               // 32B load
        v8h a0 = *(const v8h*)(pwp + kb);                       // tile 0
        v8h a1 = *(const v8h*)(pwp + kb + 8);
#pragma unroll
        for (int t = 0; t < 16; ++t) {
            v8h a0n = a0, a1n = a1;
            if (t < 15) {                                       // prefetch t+1
                a0n = *(const v8h*)(pwp + (t + 1) * 16 * CC + kb);
                a1n = *(const v8h*)(pwp + (t + 1) * 16 * CC + kb + 8);
            }
            v16h a;
#pragma unroll
            for (int i = 0; i < 8; ++i) { a[i] = a0[i]; a[i + 8] = a1[i]; }
            acc[t] = __builtin_amdgcn_wmma_f32_16x16x32_f16(
                false, a, false, bf, (short)0, acc[t], false, false);
            a0 = a0n; a1 = a1n;
        }
    }

    const float sv   = s[p];
    const int   nimg = p / HW;
    const int   hw   = p % HW;
    float* op = out + nimg * C_IN * HW + hw;

#pragma unroll
    for (int t = 0; t < 16; ++t) {
#pragma unroll
        for (int j = 0; j < 8; ++j) {
            const int c = t * 16 + j + (half << 3);
            float v = acc[t][j] * bnsc[c] + bnsh[c];                     // folded BN
            v = 0.5f * v * (1.0f + erff(v * 0.70710678118654752f));      // exact GELU
            op[c * HW] = v * sv;                                         // coalesced
        }
    }
}

// ---------------------------------------------------------------------------
extern "C" void kernel_launch(void* const* d_in, const int* in_sizes, int n_in,
                              void* d_out, int out_size, void* d_ws, size_t ws_size,
                              hipStream_t stream)
{
    (void)in_sizes; (void)n_in; (void)out_size; (void)ws_size;
    const float* x      = (const float*)d_in[0];
    const float* dnet_w = (const float*)d_in[1];
    const float* dnet_b = (const float*)d_in[2];
    const float* proj_w = (const float*)d_in[3];
    const float* bn_g   = (const float*)d_in[4];
    const float* bn_b   = (const float*)d_in[5];
    const float* bn_m   = (const float*)d_in[6];
    const float* bn_v   = (const float*)d_in[7];
    float* out = (float*)d_out;

    // Workspace layout (all vector-load bases 32B-aligned):
    float*    lg    = (float*)d_ws;                  // [P_TOT][88]  f32  11.9 MB
    float*    s     = lg + (size_t)P_TOT * D_BINS;   // [P_TOT]      f32
    float*    biasp = s + P_TOT;                     // [224]        f32
    float*    bnsc  = biasp + O_PAD;                 // [256]        f32
    float*    bnsh  = bnsc + C_IN;                   // [256]        f32
    _Float16* tf16  = (_Float16*)(bnsh + C_IN);      // [P_TOT][128] f16  8.65 MB
    _Float16* wf16  = tf16 + (size_t)P_TOT * CC;     // [224][256]   f16
    _Float16* pw16  = wf16 + O_PAD * C_IN;           // [256][128]   f16

    k_prep<<<64, 256, 0, stream>>>(dnet_w, dnet_b, proj_w, bn_g, bn_b, bn_m, bn_v,
                                   wf16, biasp, pw16, bnsc, bnsh);
    // 2112 M-tiles, 4 waves/block
    k_gemm1<<<528, 128, 0, stream>>>(x, wf16, biasp, lg, tf16);
    // one wave per position
    k_depth<<<P_TOT / 8, 256, 0, stream>>>(lg, out, s);
    // 2112 P-tiles, 4 waves/block
    k_gemm2<<<528, 128, 0, stream>>>(tf16, pw16, bnsc, bnsh, s, out);
}
